// Dan_attention_32212254720607
// MI455X (gfx1250) — compile-verified, matrix operand order checked by reading, not run on hardware
//
#include <hip/hip_runtime.h>
#include <hip/hip_bf16.h>
#include <stdint.h>

// ---------------------------------------------------------------------------
// Attention pipeline for MI455X (gfx1250, wave32, WMMA + TDM):
//   Xb      = bf16(x)                          (cvt kernel)
//   Wt{k,v,q} = bf16(W^T)                      (cvt+transpose kernel)
//   Q,K     = Xb @ Wt^T        bf16 row-major  (gemm_nt, OUT=1)
//   Vt      = (Xb @ Wvt^T)^T   bf16 [D][N]     (gemm_nt, OUT=2, transposed store)
//   S       = Q @ K^T          f32             (gemm_nt, OUT=0)
//   P       = softmax_rows(S)  bf16            (softmax kernel)
//   out     = P @ Vt^T         f32             (gemm_nt, OUT=0)
// Every GEMM is NT-form (both operands K-major): tiles are fed by the Tensor
// Data Mover (tensor_load_to_lds) with HW LDS row padding (64B rows + 16B pad
// == the 40-element padded LDS stride), double buffered via TENSORcnt.
// Matrix math: v_wmma_f32_16x16x32_bf16, f32 accumulate. Workspace ~536 MB.
// ---------------------------------------------------------------------------

typedef __bf16 bf16_t;
typedef __attribute__((ext_vector_type(16))) __bf16 v16bf;
typedef __attribute__((ext_vector_type(8)))  float  v8f;
typedef __attribute__((ext_vector_type(4)))  unsigned int v4u;
typedef __attribute__((ext_vector_type(8)))  int  v8i_t;
typedef __attribute__((ext_vector_type(4)))  int  v4i_t;

constexpr int NTOK   = 8192;
constexpr int DMODEL = 2048;

static __device__ __forceinline__ bf16_t f2bf(float f) {
    union { float f; uint32_t u; } in; in.f = f;
    uint32_t u = in.u;
    u += 0x7FFFu + ((u >> 16) & 1u);            // round-to-nearest-even
    uint16_t h = (uint16_t)(u >> 16);
    union { uint16_t u; bf16_t b; } out; out.u = h;
    return out.b;
}

// ---------------- TDM: 2D bf16 tile -> LDS (with HW row padding) -----------
// Loads tile_d1 rows of tile_d0 bf16 elements (row stride d0_stride elements)
// into LDS at lds_off. pad: every 16 DWORDs (64B = one 32-elem row) insert
// 4 DWORDs (16B = 8 elems) -> LDS row stride = 40 elements.
static __device__ __forceinline__ void tdm_load_tile(
    const bf16_t* gptr, uint32_t lds_off,
    uint32_t tile_d0, uint32_t tile_d1, uint32_t d0_stride)
{
    uint64_t ga = (uint64_t)(uintptr_t)gptr;
    v4u g0;
    g0[0] = 1u;                                          // count=1, user desc
    g0[1] = lds_off;                                     // lds_addr
    g0[2] = (uint32_t)(ga & 0xFFFFFFFFu);                // global_addr[31:0]
    g0[3] = (uint32_t)((ga >> 32) & 0x01FFFFFFu)         // global_addr[56:32]
          | (2u << 30);                                  // type=2 (image)
    v8i_t g1;
    g1[0] = (int)((1u << 16)                             // data_size = 2 bytes
                | (1u << 20)                             // pad_enable
                | (3u << 22)                             // pad_interval: 16 DW
                | (3u << 25));                           // pad_amount: 4 DW
    g1[1] = (int)((tile_d0 & 0xFFFFu) << 16);            // tensor_dim0[15:0]
    g1[2] = (int)(((tile_d0 >> 16) & 0xFFFFu)            // tensor_dim0[31:16]
                | ((tile_d1 & 0xFFFFu) << 16));          // tensor_dim1[15:0]
    g1[3] = (int)(((tile_d1 >> 16) & 0xFFFFu)            // tensor_dim1[31:16]
                | ((tile_d0 & 0xFFFFu) << 16));          // tile_dim0
    g1[4] = (int)(tile_d1 & 0xFFFFu);                    // tile_dim1 (dim2=0)
    g1[5] = (int)d0_stride;                              // dim0_stride[31:0]
    g1[6] = 0;                                           // stride[47:32], d1s
    g1[7] = 0;
    v4i_t z = {0, 0, 0, 0};
#if __clang_major__ >= 23
    v8i_t z8 = {0, 0, 0, 0, 0, 0, 0, 0};
    __builtin_amdgcn_tensor_load_to_lds(g0, g1, z, z, z8, 0);
#else
    __builtin_amdgcn_tensor_load_to_lds(g0, g1, z, z, 0);
#endif
}

// ------------------------------- f32 -> bf16 -------------------------------
__global__ __launch_bounds__(256)
void cvt_f32_to_bf16(const float* __restrict__ src, bf16_t* __restrict__ dst, int n) {
    int i      = blockIdx.x * blockDim.x + threadIdx.x;
    int stride = gridDim.x * blockDim.x;
    for (int c = i; c * 4 < n; c += stride) {
        int j = c * 4;
        float4 f = *(const float4*)(src + j);
        union { ushort4 s4; bf16_t b[4]; } o;
        o.b[0] = f2bf(f.x); o.b[1] = f2bf(f.y); o.b[2] = f2bf(f.z); o.b[3] = f2bf(f.w);
        *(ushort4*)(dst + j) = o.s4;
    }
}

// --------------------- f32 [R][C] -> bf16 transpose [C][R] -----------------
__global__ __launch_bounds__(256)
void cvt_transpose_bf16(const float* __restrict__ src, bf16_t* __restrict__ dst,
                        int R, int C) {
    __shared__ float t[32][33];
    const int tx = threadIdx.x & 31, ty = threadIdx.x >> 5;   // 32 x 8
    const int r0 = blockIdx.y * 32, c0 = blockIdx.x * 32;
    #pragma unroll
    for (int i = 0; i < 4; ++i)
        t[ty + i * 8][tx] = src[(size_t)(r0 + ty + i * 8) * C + c0 + tx];
    __syncthreads();
    #pragma unroll
    for (int i = 0; i < 4; ++i)
        dst[(size_t)(c0 + ty + i * 8) * R + r0 + tx] = f2bf(t[tx][ty + i * 8]);
}

// ------------------------- bf16 NT GEMM (WMMA + TDM) -----------------------
// C = A @ B^T.  A: [M x K] row-major bf16, B: [Ncols x K] row-major bf16.
// OUT_MODE: 0 = f32 [M x Ncols], 1 = bf16 [M x Ncols], 2 = bf16 [Ncols x M].
union Frag16 { v16bf v; uint4 q[2]; };

template <int OUT_MODE>
__global__ __launch_bounds__(256)
void gemm_nt_wmma(const bf16_t* __restrict__ A,
                  const bf16_t* __restrict__ B,
                  void* __restrict__ Cout,
                  int M, int Ncols, int K)
{
    constexpr int TM = 128, TN = 128, TK = 32;
    constexpr int LDK = TK + 8;                 // == TDM padded row stride
    __shared__ bf16_t As[2][TM][LDK];
    __shared__ bf16_t Bs[2][TN][LDK];

    const int tid  = threadIdx.x;
    const int lane = tid & 31;
    const int wave = tid >> 5;
    const int wm   = wave & 3;                  // 4 wave-rows (32 M each)
    const int wn   = wave >> 2;                 // 2 wave-cols (64 N each)

    const int bm = blockIdx.y * TM;
    const int bn = blockIdx.x * TN;

    v8f acc[2][4];
    #pragma unroll
    for (int mi = 0; mi < 2; ++mi)
        #pragma unroll
        for (int ni = 0; ni < 4; ++ni)
            #pragma unroll
            for (int e = 0; e < 8; ++e) acc[mi][ni][e] = 0.0f;

    const uint32_t ldsA = (uint32_t)(uintptr_t)(void*)&As[0][0][0];
    const uint32_t ldsB = (uint32_t)(uintptr_t)(void*)&Bs[0][0][0];
    constexpr uint32_t BUFB = (uint32_t)(TM * LDK * 2);   // bytes per buffer

    auto issue = [&](int buf, int kt) {        // one wave issues both tiles
        const int k0 = kt * TK;
        tdm_load_tile(&A[(size_t)bm * K + k0], ldsA + (uint32_t)buf * BUFB,
                      TK, TM, (uint32_t)K);
        tdm_load_tile(&B[(size_t)bn * K + k0], ldsB + (uint32_t)buf * BUFB,
                      TK, TN, (uint32_t)K);
    };

    const int kNT = K / TK;
    if (wave == 0) issue(0, 0);

    const int frow = lane & 15;                 // row/col within 16x16 frag
    const int koff = (lane >> 4) * 8;           // K-half per 16-bit A/B layout

    for (int kt = 0; kt < kNT; ++kt) {
        const int cur = kt & 1, nxt = cur ^ 1;
        const bool has_next = (kt + 1 < kNT);
        if (wave == 0) {
            if (has_next) {
                issue(nxt, kt + 1);                       // TENSORcnt: +2 -> 4
                __builtin_amdgcn_s_wait_tensorcnt(2);     // cur pair complete
            } else {
                __builtin_amdgcn_s_wait_tensorcnt(0);
            }
        }
        __syncthreads();                                  // cur tile visible

        Frag16 a[2], b[4];
        #pragma unroll
        for (int mi = 0; mi < 2; ++mi) {
            const bf16_t* p = &As[cur][wm * 32 + mi * 16 + frow][koff];
            a[mi].q[0] = *(const uint4*)p;
            a[mi].q[1] = *(const uint4*)(p + 16);
        }
        #pragma unroll
        for (int ni = 0; ni < 4; ++ni) {
            const bf16_t* p = &Bs[cur][wn * 64 + ni * 16 + frow][koff];
            b[ni].q[0] = *(const uint4*)p;
            b[ni].q[1] = *(const uint4*)(p + 16);
        }
        #pragma unroll
        for (int mi = 0; mi < 2; ++mi)
            #pragma unroll
            for (int ni = 0; ni < 4; ++ni)
                acc[mi][ni] = __builtin_amdgcn_wmma_f32_16x16x32_bf16(
                    false, a[mi].v, false, b[ni].v,
                    (short)0, acc[mi][ni], false, false);

        __syncthreads();                    // all reads of cur done before its
    }                                       // buffer is re-targeted next iter

    // epilogue: C frag -> VGPR e: lanes 0-15 M=e, lanes 16-31 M=8+e; N=lane&15
    const int crow0 = bm + wm * 32 + ((lane >> 4) * 8);
    const int ccol0 = bn + wn * 64 + (lane & 15);
    #pragma unroll
    for (int mi = 0; mi < 2; ++mi)
        #pragma unroll
        for (int ni = 0; ni < 4; ++ni)
            #pragma unroll
            for (int e = 0; e < 8; ++e) {
                const int rr = crow0 + mi * 16 + e;
                const int cc = ccol0 + ni * 16;
                if constexpr (OUT_MODE == 0)
                    ((float*)Cout)[(size_t)rr * Ncols + cc] = acc[mi][ni][e];
                else if constexpr (OUT_MODE == 1)
                    ((bf16_t*)Cout)[(size_t)rr * Ncols + cc] = f2bf(acc[mi][ni][e]);
                else
                    ((bf16_t*)Cout)[(size_t)cc * M + rr] = f2bf(acc[mi][ni][e]);
            }
}

// ------------------------------ row softmax --------------------------------
__global__ __launch_bounds__(256)
void softmax_rows(const float* __restrict__ S, bf16_t* __restrict__ P, int ncols)
{
    __shared__ float redm[8];
    __shared__ float reds[8];
    const int row  = blockIdx.x;
    const int tid  = threadIdx.x;
    const int lane = tid & 31;
    const int wave = tid >> 5;
    const float* s = S + (size_t)row * ncols;
    bf16_t*      p = P + (size_t)row * ncols;

    float m = -3.402823466e+38f;
    for (int c = tid; c < ncols; c += 256) m = fmaxf(m, s[c]);
    #pragma unroll
    for (int off = 16; off; off >>= 1) m = fmaxf(m, __shfl_xor(m, off, 32));
    if (lane == 0) redm[wave] = m;
    __syncthreads();
    float m_all = redm[0];
    #pragma unroll
    for (int i = 1; i < 8; ++i) m_all = fmaxf(m_all, redm[i]);

    float l = 0.0f;
    for (int c = tid; c < ncols; c += 256) l += __expf(s[c] - m_all);
    #pragma unroll
    for (int off = 16; off; off >>= 1) l += __shfl_xor(l, off, 32);
    if (lane == 0) reds[wave] = l;
    __syncthreads();
    float l_all = 0.0f;
    #pragma unroll
    for (int i = 0; i < 8; ++i) l_all += reds[i];
    const float inv = 1.0f / l_all;

    for (int c = tid; c < ncols; c += 256) p[c] = f2bf(__expf(s[c] - m_all) * inv);
}

// ------------------------------- launcher ----------------------------------
extern "C" void kernel_launch(void* const* d_in, const int* in_sizes, int n_in,
                              void* d_out, int out_size, void* d_ws, size_t ws_size,
                              hipStream_t stream)
{
    const float* x  = (const float*)d_in[0];
    const float* wk = (const float*)d_in[1];
    const float* wv = (const float*)d_in[2];
    const float* wq = (const float*)d_in[3];
    float* out = (float*)d_out;

    char* w = (char*)d_ws;
    auto alloc = [&](size_t bytes) -> void* {
        void* p = (void*)w;
        w += (bytes + 255) & ~(size_t)255;
        return p;
    };
    bf16_t* Xb   = (bf16_t*)alloc((size_t)NTOK * DMODEL * 2);
    bf16_t* Wkt  = (bf16_t*)alloc((size_t)DMODEL * DMODEL * 2);
    bf16_t* Wvt  = (bf16_t*)alloc((size_t)DMODEL * DMODEL * 2);
    bf16_t* Wqt  = (bf16_t*)alloc((size_t)DMODEL * DMODEL * 2);
    bf16_t* Qb   = (bf16_t*)alloc((size_t)NTOK * DMODEL * 2);
    bf16_t* Kb   = (bf16_t*)alloc((size_t)NTOK * DMODEL * 2);
    bf16_t* Vt   = (bf16_t*)alloc((size_t)NTOK * DMODEL * 2);   // [DMODEL][NTOK]
    float*  S    = (float*)alloc((size_t)NTOK * NTOK * 4);
    bf16_t* P    = (bf16_t*)alloc((size_t)NTOK * NTOK * 2);

    // 1) convert x; convert + transpose weights
    {
        int nx = NTOK * DMODEL;
        cvt_f32_to_bf16<<<(nx / 4 + 255) / 256, 256, 0, stream>>>(x, Xb, nx);
        dim3 gt(DMODEL / 32, DMODEL / 32);
        cvt_transpose_bf16<<<gt, 256, 0, stream>>>(wk, Wkt, DMODEL, DMODEL);
        cvt_transpose_bf16<<<gt, 256, 0, stream>>>(wv, Wvt, DMODEL, DMODEL);
        cvt_transpose_bf16<<<gt, 256, 0, stream>>>(wq, Wqt, DMODEL, DMODEL);
    }

    // 2) projections: Q,K row-major bf16; V stored transposed [DMODEL][NTOK]
    dim3 gproj(DMODEL / 128, NTOK / 128);
    gemm_nt_wmma<1><<<gproj, 256, 0, stream>>>(Xb, Wqt, Qb, NTOK, DMODEL, DMODEL);
    gemm_nt_wmma<1><<<gproj, 256, 0, stream>>>(Xb, Wkt, Kb, NTOK, DMODEL, DMODEL);
    gemm_nt_wmma<2><<<gproj, 256, 0, stream>>>(Xb, Wvt, Vt, NTOK, DMODEL, DMODEL);

    // 3) S = Q @ K^T (f32)
    dim3 gs(NTOK / 128, NTOK / 128);
    gemm_nt_wmma<0><<<gs, 256, 0, stream>>>(Qb, Kb, S, NTOK, NTOK, DMODEL);

    // 4) P = softmax(S) rows (bf16)
    softmax_rows<<<NTOK, 256, 0, stream>>>(S, P, NTOK);

    // 5) out = P @ Vt^T (f32)
    dim3 go(DMODEL / 128, NTOK / 128);
    gemm_nt_wmma<0><<<go, 256, 0, stream>>>(P, Vt, out, NTOK, DMODEL, NTOK);
}
